// DeorphaNN_75746043232846
// MI455X (gfx1250) — matrix operand order repaired and checked by reading, not compiled, not run on hardware
//
#include <hip/hip_runtime.h>
#include <hip/hip_bf16.h>
#include <math.h>

// ---------------- problem constants ----------------
#define NN    50000      // nodes
#define EE    400000     // edges
#define FIN   128        // input feature dim
#define HH    10         // heads
#define CC    32         // per-head channels
#define HC    320        // HH*CC
#define GG    64         // graphs
#define NEG_SLOPE 0.2f
#define BN_EPS    1e-5f

#define LDS_PITCH (FIN + 8)   // halves per row; +8 halves (16B) kills bank conflicts

typedef __attribute__((ext_vector_type(16))) _Float16 v16h;
typedef __attribute__((ext_vector_type(8)))  float    v8f;

// ---------------- WMMA fragment helpers (gfx1250 wave32 layouts) ----------------
// A: 16x32 f16 (M x K), staged in LDS as f16 rows of LDS_PITCH halves.
// lanes 0-15:  M=lane,    K = k0 + {0..7, 16..23}
// lanes 16-31: M=lane-16, K = k0 + {8..15, 24..31}
__device__ __forceinline__ v16h load_a_frag_lds(const _Float16* __restrict__ lds,
                                                int k0, int lane) {
    const int m  = lane & 15;
    const int kb = k0 + ((lane & 16) ? 8 : 0);
    const _Float16* row = lds + m * LDS_PITCH;
    v16h a;
#pragma unroll
    for (int j = 0; j < 8; ++j) a[j]     = row[kb + j];
#pragma unroll
    for (int j = 0; j < 8; ++j) a[8 + j] = row[kb + 16 + j];
    return a;
}

// B: 32x16 f16 (K x N), B[k][n] = W[n][k], W row-major [HC, FIN] (L2-resident).
// lanes 0-15: N=lane, K = k0+{0..7,16..23} ; lanes 16-31: N=lane-16, K = k0+{8..15,24..31}
__device__ __forceinline__ v16h load_b_frag(const float* __restrict__ W, int ldw,
                                            int n0, int k0, int lane) {
    const int n  = n0 + (lane & 15);
    const int kb = k0 + ((lane & 16) ? 8 : 0);
    const float* row = W + (size_t)n * ldw + kb;
    v16h b;
#pragma unroll
    for (int j = 0; j < 16; ++j) b[j] = (_Float16)row[(j < 8) ? j : j + 8];
    return b;
}

__device__ __forceinline__ v8f wmma_step(v16h a, v16h b, v8f c) {
    return __builtin_amdgcn_wmma_f32_16x16x32_f16(false, a, false, b,
                                                  (short)0, c, false, false);
}

// stage a 16xFIN f32 tile into LDS as f16 (coalesced global reads)
__device__ __forceinline__ void stage_a_tile(const float* __restrict__ A,
                                             int m0, _Float16* __restrict__ lds) {
    for (int i = threadIdx.x; i < 16 * FIN; i += blockDim.x) {
        const int m = i >> 7;           // FIN == 128
        const int k = i & (FIN - 1);
        lds[m * LDS_PITCH + k] = (_Float16)A[(size_t)(m0 + m) * FIN + k];
    }
    __syncthreads();
}

// ---------------- utility kernels ----------------
__global__ void fill_f32(float* __restrict__ p, long long n, float v) {
    long long i = (long long)blockIdx.x * blockDim.x + threadIdx.x;
    long long stride = (long long)gridDim.x * blockDim.x;
    for (; i < n; i += stride) p[i] = v;
}

__global__ void batchnorm_kernel(const float* __restrict__ x,
                                 const float* __restrict__ g, const float* __restrict__ b,
                                 const float* __restrict__ mu, const float* __restrict__ var,
                                 float* __restrict__ xn) {
    long long i = (long long)blockIdx.x * blockDim.x + threadIdx.x;
    if (i >= (long long)NN * FIN) return;
    int f = (int)(i & (FIN - 1));
    xn[i] = (x[i] - mu[f]) * rsqrtf(var[f] + BN_EPS) * g[f] + b[f];
}

// ---------------- node-side GEMMs: xl = xn*Wl^T + bl ; xr = xn*Wr^T + br ----------------
// grid (NN/16, 5), block 256 (8 waves); A tile shared via LDS; 40 column tiles.
__global__ void node_gemm_kernel(const float* __restrict__ xn,
                                 const float* __restrict__ Wl, const float* __restrict__ bl,
                                 const float* __restrict__ Wr, const float* __restrict__ br,
                                 float* __restrict__ xl, float* __restrict__ xr) {
    __shared__ _Float16 aTile[16 * LDS_PITCH];
    const int wave = threadIdx.x >> 5;
    const int lane = threadIdx.x & 31;
    const int m0   = blockIdx.x * 16;
    const int t    = blockIdx.y * 8 + wave;            // 0..39, exact (no idle waves)

    stage_a_tile(xn, m0, aTile);

    const float* W; const float* bias; float* out; int n0;
    if (t < 20) { W = Wl; bias = bl; out = xl; n0 = t * 16; }
    else        { W = Wr; bias = br; out = xr; n0 = (t - 20) * 16; }

    v8f acc = {};
#pragma unroll
    for (int k0 = 0; k0 < FIN; k0 += 32) {
        v16h a = load_a_frag_lds(aTile, k0, lane);
        v16h b = load_b_frag(W, FIN, n0, k0, lane);
        acc = wmma_step(a, b, acc);
    }
    const int n  = n0 + (lane & 15);
    const float bv = bias[n];
    const int mb = m0 + ((lane & 16) ? 8 : 0);
#pragma unroll
    for (int r = 0; r < 8; ++r)
        out[(size_t)(mb + r) * HC + n] = acc[r] + bv;
}

// ---------------- fused edge kernel ----------------
// ea tile [16 edges x 16 cols] via WMMA (never materialized globally), then
// m = ea + xl[src] + xr[dst] ; lrelu ; * att ; reduce over cols -> partial alpha.
// block 160 (5 waves), grid (EE/16, 4): 20 column tiles exactly, no idle waves.
__global__ void edge_alpha_kernel(const float* __restrict__ edge_attr,
                                  const float* __restrict__ We,
                                  const float* __restrict__ xl,
                                  const float* __restrict__ xr,
                                  const float* __restrict__ att,
                                  const int*  __restrict__ src,
                                  const int*  __restrict__ dst,
                                  float* __restrict__ alpha) {
    __shared__ _Float16 aTile[16 * LDS_PITCH];
    const int wave = threadIdx.x >> 5;
    const int lane = threadIdx.x & 31;
    const int t    = blockIdx.y * 5 + wave;            // column tile 0..19
    const int e0 = blockIdx.x * 16;
    const int n0 = t * 16;

    // prefetch next edge tile's rows (global_prefetch_b8)
    if (e0 + 16 < EE)
        __builtin_prefetch(edge_attr + (size_t)(e0 + 16 + (lane & 15)) * FIN, 0, 0);

    stage_a_tile(edge_attr, e0, aTile);

    v8f acc = {};
#pragma unroll
    for (int k0 = 0; k0 < FIN; k0 += 32) {
        v16h a = load_a_frag_lds(aTile, k0, lane);
        v16h b = load_b_frag(We, FIN, n0, k0, lane);
        acc = wmma_step(a, b, acc);
    }

    const int h   = n0 >> 5;                           // head of this tile
    const int n   = n0 + (lane & 15);                  // global column
    const int cIn = n & (CC - 1);                      // channel within head
    const float av = att[h * CC + cIn];
    const int eb  = e0 + ((lane & 16) ? 8 : 0);

    float s[8];
#pragma unroll
    for (int r = 0; r < 8; ++r) {
        const int e  = eb + r;
        const int si = src[e], di = dst[e];
        float m = acc[r] + xl[(size_t)si * HC + n] + xr[(size_t)di * HC + n];
        float lr = (m > 0.0f) ? m : NEG_SLOPE * m;
        s[r] = lr * av;
    }
    // reduce over the 16 columns (lanes 0-15 and 16-31 reduce independently)
#pragma unroll
    for (int mask = 1; mask < 16; mask <<= 1) {
#pragma unroll
        for (int r = 0; r < 8; ++r) s[r] += __shfl_xor(s[r], mask, 32);
    }
    if ((lane & 15) == 0) {
#pragma unroll
        for (int r = 0; r < 8; ++r)
            atomicAdd(&alpha[(size_t)(eb + r) * HH + h], s[r]);
    }
}

// ---------------- segment softmax ----------------
__device__ __forceinline__ void atomicMaxF32(float* addr, float v) {
    if (v >= 0.0f) atomicMax((int*)addr, __float_as_int(v));
    else           atomicMin((unsigned int*)addr, __float_as_uint(v));
}

__global__ void seg_max_kernel(const float* __restrict__ alpha,
                               const int* __restrict__ dst,
                               float* __restrict__ amax) {
    long long i = (long long)blockIdx.x * blockDim.x + threadIdx.x;
    if (i >= (long long)EE * HH) return;
    int e = (int)(i / HH), h = (int)(i - (long long)e * HH);
    atomicMaxF32(&amax[(size_t)dst[e] * HH + h], alpha[i]);
}

__global__ void seg_expsum_kernel(float* __restrict__ alpha,
                                  const int* __restrict__ dst,
                                  const float* __restrict__ amax,
                                  float* __restrict__ denom) {
    long long i = (long long)blockIdx.x * blockDim.x + threadIdx.x;
    if (i >= (long long)EE * HH) return;
    int e = (int)(i / HH), h = (int)(i - (long long)e * HH);
    size_t dh = (size_t)dst[e] * HH + h;
    float v = __expf(alpha[i] - amax[dh]);
    alpha[i] = v;
    atomicAdd(&denom[dh], v);
}

// normalize alpha in place: alpha <- alpha / (denom[dst,h]+eps) / H
__global__ void seg_norm_kernel(float* __restrict__ alpha,
                                const int* __restrict__ dst,
                                const float* __restrict__ denom) {
    long long i = (long long)blockIdx.x * blockDim.x + threadIdx.x;
    if (i >= (long long)EE * HH) return;
    int e = (int)(i / HH), h = (int)(i - (long long)e * HH);
    alpha[i] = alpha[i] / (denom[(size_t)dst[e] * HH + h] + 1e-16f) * (1.0f / HH);
}

// ---------------- weighted scatter aggregation ----------------
// thread per (e,c); sum over the 10 heads in registers -> one atomic per (e,c)
__global__ void aggregate_kernel(const float* __restrict__ alpha,   // normalized
                                 const float* __restrict__ xl,
                                 const int* __restrict__ src,
                                 const int* __restrict__ dst,
                                 float* __restrict__ nacc) {
    long long i = (long long)blockIdx.x * blockDim.x + threadIdx.x;
    if (i >= (long long)EE * CC) return;
    int c = (int)(i & (CC - 1));
    int e = (int)(i >> 5);
    const int si = src[e], di = dst[e];
    const float* xls = xl + (size_t)si * HC + c;
    const float* al  = alpha + (size_t)e * HH;
    float acc = 0.0f;
#pragma unroll
    for (int h = 0; h < HH; ++h)
        acc += al[h] * xls[h * CC];
    atomicAdd(&nacc[(size_t)di * CC + c], acc);
}

// ---------------- bias + relu + global mean pool ----------------
__global__ void node_pool_kernel(const float* __restrict__ nacc,
                                 const float* __restrict__ bias_out,
                                 const int* __restrict__ batch,
                                 float* __restrict__ pooled,
                                 float* __restrict__ cnt) {
    long long i = (long long)blockIdx.x * blockDim.x + threadIdx.x;
    if (i >= (long long)NN * CC) return;
    int n = (int)(i >> 5), c = (int)(i & (CC - 1));
    float v = nacc[i] + bias_out[c];
    v = fmaxf(v, 0.0f);
    int g = batch[n];
    atomicAdd(&pooled[(size_t)g * CC + c], v);
    if (c == 0) atomicAdd(&cnt[g], 1.0f);
}

// ---------------- final classifier [G,2] ----------------
__global__ void classify_kernel(const float* __restrict__ pooled,
                                const float* __restrict__ cnt,
                                const float* __restrict__ Wlin,
                                const float* __restrict__ blin,
                                float* __restrict__ out) {
    int i = blockIdx.x * blockDim.x + threadIdx.x;
    if (i >= GG * 2) return;
    int g = i >> 1, j = i & 1;
    float cn = fmaxf(cnt[g], 1.0f);
    float s = blin[j];
#pragma unroll
    for (int c = 0; c < CC; ++c)
        s += (pooled[(size_t)g * CC + c] / cn) * Wlin[j * CC + c];
    out[i] = s;
}

// ---------------- host launcher ----------------
extern "C" void kernel_launch(void* const* d_in, const int* in_sizes, int n_in,
                              void* d_out, int out_size, void* d_ws, size_t ws_size,
                              hipStream_t stream) {
    const float* x         = (const float*)d_in[0];
    const float* edge_attr = (const float*)d_in[1];
    const float* bn_gamma  = (const float*)d_in[2];
    const float* bn_beta   = (const float*)d_in[3];
    const float* bn_mean   = (const float*)d_in[4];
    const float* bn_var    = (const float*)d_in[5];
    const float* Wl        = (const float*)d_in[6];
    const float* bl        = (const float*)d_in[7];
    const float* Wr        = (const float*)d_in[8];
    const float* br        = (const float*)d_in[9];
    const float* We        = (const float*)d_in[10];
    const float* att       = (const float*)d_in[11];
    const float* bias_out  = (const float*)d_in[12];
    const float* Wlin      = (const float*)d_in[13];
    const float* blin      = (const float*)d_in[14];
    const int*   eidx      = (const int*)d_in[15];   // [2,E]
    const int*   batch     = (const int*)d_in[16];   // [N]
    const int*   src = eidx;
    const int*   dst = eidx + EE;
    float* out = (float*)d_out;

    // workspace carve-up (floats)
    float* ws    = (float*)d_ws;
    float* xn    = ws;                                  // N*FIN
    float* xl    = xn    + (size_t)NN * FIN;            // N*HC
    float* xr    = xl    + (size_t)NN * HC;             // N*HC
    float* alpha = xr    + (size_t)NN * HC;             // E*HH
    float* amax  = alpha + (size_t)EE * HH;             // N*HH
    float* denom = amax  + (size_t)NN * HH;             // N*HH
    float* nacc  = denom + (size_t)NN * HH;             // N*CC
    float* pooled= nacc  + (size_t)NN * CC;             // G*CC
    float* cnt   = pooled+ (size_t)GG * CC;             // G

    const int TB = 256;

    // init accumulators (alpha=0, amax=-inf, denom/nacc/pooled/cnt=0, all contiguous)
    fill_f32<<<2048, TB, 0, stream>>>(alpha, (long long)EE * HH, 0.0f);
    fill_f32<<<2048, TB, 0, stream>>>(amax,  (long long)NN * HH, -INFINITY);
    fill_f32<<<2048, TB, 0, stream>>>(denom,
        (long long)NN * HH + (long long)NN * CC + (long long)GG * CC + GG, 0.0f);

    // batchnorm
    batchnorm_kernel<<<(NN * FIN + TB - 1) / TB, TB, 0, stream>>>(
        x, bn_gamma, bn_beta, bn_mean, bn_var, xn);

    // node GEMMs (WMMA, LDS-staged A tile)
    node_gemm_kernel<<<dim3(NN / 16, 5), TB, 0, stream>>>(xn, Wl, bl, Wr, br, xl, xr);

    // fused edge GEMM + GATv2 scoring (WMMA, LDS-staged A tile)
    edge_alpha_kernel<<<dim3(EE / 16, 4), 160, 0, stream>>>(
        edge_attr, We, xl, xr, att, src, dst, alpha);

    // segment softmax
    long long ehTot = (long long)EE * HH;
    unsigned ehBlocks = (unsigned)((ehTot + TB - 1) / TB);
    seg_max_kernel<<<ehBlocks, TB, 0, stream>>>(alpha, dst, amax);
    seg_expsum_kernel<<<ehBlocks, TB, 0, stream>>>(alpha, dst, amax, denom);
    seg_norm_kernel<<<ehBlocks, TB, 0, stream>>>(alpha, dst, denom);

    // weighted scatter aggregation (heads reduced in registers)
    long long aggTot = (long long)EE * CC;
    aggregate_kernel<<<(unsigned)((aggTot + TB - 1) / TB), TB, 0, stream>>>(
        alpha, xl, src, dst, nacc);

    // relu + bias + per-graph mean pool
    node_pool_kernel<<<(NN * CC + TB - 1) / TB, TB, 0, stream>>>(
        nacc, bias_out, batch, pooled, cnt);

    // classifier
    classify_kernel<<<1, 128, 0, stream>>>(pooled, cnt, Wlin, blin, out);

    (void)in_sizes; (void)n_in; (void)out_size; (void)ws_size;
}
